// DirGCNConv_2_27152783245348
// MI455X (gfx1250) — matrix operand order repaired and checked by reading, not compiled, not run on hardware
//
#include <hip/hip_runtime.h>
#include <hip/hip_bf16.h>

// ---------------------------------------------------------------------------
// DirGCNConv forward for MI455X (gfx1250, wave32).
//   - SpMM passes: edge-parallel, one wave32 per edge, 2 feats/lane (float2),
//     relaxed agent-scope f32 atomics -> global_atomic_add_f32 in L2.
//   - Dense 64x64 linears: bf16 WMMA (v_wmma_f32_16x16x32_bf16).
//     W is transposed+converted to bf16 in LDS once per block; B fragments
//     come from LDS via ds_load_b128; A fragments via global float4 loads.
//     Each wave computes 4 16x16 tiles (8 row-tiles x 4 col-tiles / block).
// ---------------------------------------------------------------------------

typedef __attribute__((ext_vector_type(16))) __bf16 v16bf;
typedef __attribute__((ext_vector_type(8)))  float  v8f;

#define F 64              // F_IN == F_OUT == 64
#define TILES_PER_WAVE 4  // row-tiles per wave in the GEMM

__device__ __forceinline__ void atomic_add_f32(float* p, float v) {
    __hip_atomic_fetch_add(p, v, __ATOMIC_RELAXED, __HIP_MEMORY_SCOPE_AGENT);
}

// ---- degree accumulation: d_out[row]+=1, d_in[col]+=1 ----------------------
__global__ __launch_bounds__(256) void degree_kernel(
    const int* __restrict__ row, const int* __restrict__ col,
    float* __restrict__ dOut, float* __restrict__ dIn, int E)
{
    int e = blockIdx.x * blockDim.x + threadIdx.x;
    if (e >= E) return;
    atomic_add_f32(&dOut[row[e]], 1.0f);
    atomic_add_f32(&dIn[col[e]], 1.0f);
}

// ---- second-order degree vectors ------------------------------------------
// g0 = Av(d_in), g1 = Atv(d_out), g2 = Av(d_out), g3 = Atv(d_in)
__global__ __launch_bounds__(256) void secdeg_kernel(
    const int* __restrict__ row, const int* __restrict__ col,
    const float* __restrict__ dOut, const float* __restrict__ dIn,
    float* __restrict__ g0, float* __restrict__ g1,
    float* __restrict__ g2, float* __restrict__ g3, int E)
{
    int e = blockIdx.x * blockDim.x + threadIdx.x;
    if (e >= E) return;
    int r = row[e], c = col[e];
    float doR = dOut[r], doC = dOut[c], diR = dIn[r], diC = dIn[c];
    atomic_add_f32(&g0[r], diC);   // Av(d_in)
    atomic_add_f32(&g1[c], doR);   // Atv(d_out)
    atomic_add_f32(&g2[r], doC);   // Av(d_out)
    atomic_add_f32(&g3[c], diR);   // Atv(d_in)
}

// ---- in-place d -> d^{-1/2} with inf->0 (masked_fill) over 6 packed arrays -
__global__ __launch_bounds__(256) void invsqrt_kernel(float* __restrict__ d, int n)
{
    int i = blockIdx.x * blockDim.x + threadIdx.x;
    if (i >= n) return;
    float v = d[i];
    d[i] = (v > 0.0f) ? __frsqrt_rn(fmaxf(v, 1e-30f)) : 0.0f;
}

// ---- out = sum_i coef_i * b_i (bias broadcast init) ------------------------
__global__ __launch_bounds__(256) void bias_init_kernel(
    float* __restrict__ out,
    const float* __restrict__ b0, const float* __restrict__ b1,
    const float* __restrict__ b2, const float* __restrict__ b3,
    const float* __restrict__ b4, const float* __restrict__ b5,
    float c0, float c1, float c2, float c3, float c4, float c5, int total)
{
    int i = blockIdx.x * blockDim.x + threadIdx.x;
    if (i >= total) return;
    int f = i & (F - 1);
    out[i] = c0 * b0[f] + c1 * b1[f] + c2 * b2[f] + c3 * b3[f] + c4 * b4[f] + c5 * b5[f];
}

// ---- SpMM: out[dst[e]] += scale[src[e]] * xin[src[e]]  (scale nullable) ----
// One wave32 per edge, float2 per lane (64 features).
__global__ __launch_bounds__(256) void spmm_kernel(
    const int* __restrict__ src, const int* __restrict__ dst,
    const float* __restrict__ scale, const float* __restrict__ xin,
    float* __restrict__ out, int E)
{
    int gid  = blockIdx.x * blockDim.x + threadIdx.x;
    int e    = gid >> 5;
    int lane = gid & 31;
    if (e >= E) return;
    int s = src[e];
    int d = dst[e];
    float sc = scale ? scale[s] : 1.0f;
    const float2 v = *(const float2*)(xin + (size_t)s * F + lane * 2);
    float* po = out + (size_t)d * F + lane * 2;
    atomic_add_f32(po + 0, sc * v.x);
    atomic_add_f32(po + 1, sc * v.y);
}

// ---- WMMA GEMM-accumulate: out += coef * ((ascale .* h) @ W) ---------------
// h: [nRows x 64] f32, W: [64 x 64] f32 row-major (K x N), out: [nRows x 64].
// 256 threads = 8 waves. Block stages W^T as bf16 in LDS (8 KB), then each
// wave computes TILES_PER_WAVE 16x16 output tiles at its col-tile (wave&3).
// Block covers 2*TILES_PER_WAVE row-tiles x 4 col-tiles.
__global__ __launch_bounds__(256) void gemm_acc_kernel(
    const float* __restrict__ h, const float* __restrict__ ascale,
    const float* __restrict__ W, float coef, float* __restrict__ out, int nRows)
{
    __shared__ alignas(32) __bf16 Wt[F * F];   // Wt[n*64 + k] = bf16(W[k][n])

    const int tid = threadIdx.x;

    // cooperative transpose+convert: 4 passes of 256 threads x float4
#pragma unroll
    for (int base = 0; base < F * F; base += 256 * 4) {
        int idx = base + tid * 4;         // flat index k*64 + n, n % 4 == 0
        int k = idx >> 6;
        int n = idx & (F - 1);
        float4 w4 = *(const float4*)(W + idx);
        Wt[(n + 0) * F + k] = (__bf16)w4.x;
        Wt[(n + 1) * F + k] = (__bf16)w4.y;
        Wt[(n + 2) * F + k] = (__bf16)w4.z;
        Wt[(n + 3) * F + k] = (__bf16)w4.w;
    }
    __syncthreads();

    const int lane    = tid & 31;
    const int wave    = tid >> 5;
    const int halfsel = lane >> 4;        // 0: lanes 0-15, 1: lanes 16-31
    const int lrow    = lane & 15;
    const int n0      = (wave & 3) * 16;
    const int col     = n0 + lrow;        // B/C/D column owned by this lane
    const __bf16* wtcol = Wt + col * F;   // contiguous K for this column

    for (int it = 0; it < TILES_PER_WAVE; ++it) {
        int rowTile = blockIdx.x * (2 * TILES_PER_WAVE)
                    + (wave >> 2) * TILES_PER_WAVE + it;
        int m0 = rowTile * 16;
        if (m0 >= nRows) break;

        int arow  = m0 + lrow;            // A-matrix row owned by this lane
        float asc = ascale[arow];
        const float* hrow = h + (size_t)arow * F;

        v8f c = {};
#pragma unroll
        for (int kk = 0; kk < F; kk += 32) {
            // A 16x32 bf16 layout (ISA 7.12.2): lane<16 holds K 0-7,16-23;
            // lane>=16 holds K 8-15,24-31 -> two contiguous float4 pairs.
            const float4 a0 = *(const float4*)(hrow + kk + (halfsel << 3));
            const float4 a1 = *(const float4*)(hrow + kk + (halfsel << 3) + 4);
            const float4 a2 = *(const float4*)(hrow + kk + 16 + (halfsel << 3));
            const float4 a3 = *(const float4*)(hrow + kk + 16 + (halfsel << 3) + 4);
            v16bf a;
            a[0]  = (__bf16)(asc * a0.x); a[1]  = (__bf16)(asc * a0.y);
            a[2]  = (__bf16)(asc * a0.z); a[3]  = (__bf16)(asc * a0.w);
            a[4]  = (__bf16)(asc * a1.x); a[5]  = (__bf16)(asc * a1.y);
            a[6]  = (__bf16)(asc * a1.z); a[7]  = (__bf16)(asc * a1.w);
            a[8]  = (__bf16)(asc * a2.x); a[9]  = (__bf16)(asc * a2.y);
            a[10] = (__bf16)(asc * a2.z); a[11] = (__bf16)(asc * a2.w);
            a[12] = (__bf16)(asc * a3.x); a[13] = (__bf16)(asc * a3.y);
            a[14] = (__bf16)(asc * a3.z); a[15] = (__bf16)(asc * a3.w);

            // B 32x16: same K order as A: elements 0-7 hold K = kk + 8*halfsel + i,
            // elements 8-15 hold K = kk + 16 + 8*halfsel + (i-8) (2x ds_load_b128).
            typedef __attribute__((ext_vector_type(8))) __bf16 v8bf;
            const v8bf b0 = *(const v8bf*)(wtcol + kk + (halfsel << 3));
            const v8bf b1 = *(const v8bf*)(wtcol + kk + 16 + (halfsel << 3));
            v16bf b = __builtin_shufflevector(b0, b1, 0,1,2,3,4,5,6,7,8,9,10,11,12,13,14,15);

            c = __builtin_amdgcn_wmma_f32_16x16x32_bf16(
                    /*neg_a=*/false, a, /*neg_b=*/false, b,
                    /*c_mod=*/(short)0, c, /*reuse_a=*/false, /*reuse_b=*/false);
        }
        // C/D f32 16x16: VGPR r -> M = r + 8*halfsel, N = lane%16.
#pragma unroll
        for (int r = 0; r < 8; ++r) {
            int orow = m0 + r + (halfsel << 3);
            out[(size_t)orow * F + col] += coef * c[r];
        }
    }
}

// ---------------------------------------------------------------------------
extern "C" void kernel_launch(void* const* d_in, const int* in_sizes, int n_in,
                              void* d_out, int out_size, void* d_ws, size_t ws_size,
                              hipStream_t stream)
{
    const float* x    = (const float*)d_in[0];
    const int*   ei   = (const int*)d_in[1];
    const float* W_sd = (const float*)d_in[2];  const float* b_sd = (const float*)d_in[3];
    const float* W_ds = (const float*)d_in[4];  const float* b_ds = (const float*)d_in[5];
    const float* W0   = (const float*)d_in[6];  const float* b0   = (const float*)d_in[7];
    const float* W1   = (const float*)d_in[8];  const float* b1   = (const float*)d_in[9];
    const float* W2   = (const float*)d_in[10]; const float* b2   = (const float*)d_in[11];
    const float* W3   = (const float*)d_in[12]; const float* b3   = (const float*)d_in[13];
    float* out = (float*)d_out;

    const int N = in_sizes[0] / F;
    const int E = in_sizes[1] / 2;
    const int* row = ei;       // edge_index[0]
    const int* col = ei + E;   // edge_index[1]

    // scalar coefficients: (1+a)*a and (1+a)*(1-a) for alpha/beta/gamma = 0.5
    const float ALPHA = 0.5f, BETA = 0.5f, GAMA = 0.5f;
    const float c_sd = (1.0f + ALPHA) * ALPHA, c_ds = (1.0f + ALPHA) * (1.0f - ALPHA);
    const float c_0  = (1.0f + BETA)  * BETA,  c_1  = (1.0f + BETA)  * (1.0f - BETA);
    const float c_2  = (1.0f + GAMA)  * GAMA,  c_3  = (1.0f + GAMA)  * (1.0f - GAMA);

    // workspace layout (floats):
    //   [0,N)     d_out -> iso        [N,2N)   d_in  -> isi
    //   [2N,3N)   g0 = Av(d_in)  -> sAAt
    //   [3N,4N)   g1 = Atv(d_out)-> sAtA
    //   [4N,5N)   g2 = Av(d_out) -> sAAo
    //   [5N,6N)   g3 = Atv(d_in) -> sAAi
    //   [6N, 6N+64N)    hbuf        [70N, 134N)   tmp
    float* ws   = (float*)d_ws;
    float* dOut = ws;
    float* dIn  = ws + (size_t)N;
    float* g0   = ws + (size_t)2 * N;
    float* g1   = ws + (size_t)3 * N;
    float* g2   = ws + (size_t)4 * N;
    float* g3   = ws + (size_t)5 * N;
    float* hbuf = ws + (size_t)6 * N;
    float* tmp  = ws + (size_t)6 * N + (size_t)N * F;

    const int THREADS = 256;
    const int edgeBlocks  = (E + THREADS - 1) / THREADS;
    const int spmmBlocks  = (int)(((long long)E * 32 + THREADS - 1) / THREADS);
    const int nodeFBlocks = (int)(((long long)N * F + THREADS - 1) / THREADS);
    const int rowTiles    = (N + 15) / 16;
    const int gemmBlocks  = (rowTiles + 2 * TILES_PER_WAVE - 1) / (2 * TILES_PER_WAVE);
    const size_t featBytes = (size_t)N * F * sizeof(float);

    // 1) degrees + second-order degrees
    hipMemsetAsync(ws, 0, (size_t)6 * N * sizeof(float), stream);
    degree_kernel<<<edgeBlocks, THREADS, 0, stream>>>(row, col, dOut, dIn, E);
    secdeg_kernel<<<edgeBlocks, THREADS, 0, stream>>>(row, col, dOut, dIn, g0, g1, g2, g3, E);
    invsqrt_kernel<<<(6 * N + THREADS - 1) / THREADS, THREADS, 0, stream>>>(ws, 6 * N);

    // 2) out = combined bias
    bias_init_kernel<<<nodeFBlocks, THREADS, 0, stream>>>(
        out, b_sd, b_ds, b0, b1, b2, b3, c_sd, c_ds, c_0, c_1, c_2, c_3, N * F);

#define SPMM(SRC, DST, SCALE, XIN, OUT) \
    spmm_kernel<<<spmmBlocks, THREADS, 0, stream>>>((SRC), (DST), (SCALE), (XIN), (OUT), E)
#define GEMM(H, ASCALE, W, COEF) \
    gemm_acc_kernel<<<gemmBlocks, THREADS, 0, stream>>>((H), (ASCALE), (W), (COEF), out, N)

    // 3) term A_x = iso * Av(isi * x)           -> W_sd
    hipMemsetAsync(hbuf, 0, featBytes, stream);
    SPMM(col, row, dIn, x, hbuf);               // Av with src-scale isi
    GEMM(hbuf, dOut, W_sd, c_sd);               // row-scale iso fused into A

    // 4) term At_x = isi * Atv(iso * x)         -> W_ds
    hipMemsetAsync(hbuf, 0, featBytes, stream);
    SPMM(row, col, dOut, x, hbuf);              // Atv with src-scale iso
    GEMM(hbuf, dIn, W_ds, c_ds);

    // 5) term AAt_x = sAAt * Av(Atv(sAAt * x))  -> W0
    hipMemsetAsync(tmp, 0, featBytes, stream);
    hipMemsetAsync(hbuf, 0, featBytes, stream);
    SPMM(row, col, g0, x, tmp);                 // Atv(sAAt * x)
    SPMM(col, row, (const float*)nullptr, tmp, hbuf);  // Av(.)
    GEMM(hbuf, g0, W0, c_0);

    // 6) term AtA_x = sAtA * Atv(Av(sAtA * x))  -> W1
    hipMemsetAsync(tmp, 0, featBytes, stream);
    hipMemsetAsync(hbuf, 0, featBytes, stream);
    SPMM(col, row, g1, x, tmp);                 // Av(sAtA * x)
    SPMM(row, col, (const float*)nullptr, tmp, hbuf);  // Atv(.)
    GEMM(hbuf, g1, W1, c_1);

    // 7) term AA_x = sAAo * Av(Av(sAAi * x))    -> W2
    hipMemsetAsync(tmp, 0, featBytes, stream);
    hipMemsetAsync(hbuf, 0, featBytes, stream);
    SPMM(col, row, g3, x, tmp);                 // Av(sAAi * x)
    SPMM(col, row, (const float*)nullptr, tmp, hbuf);  // Av(.)
    GEMM(hbuf, g2, W2, c_2);

    // 8) term AtAt_x = sAAi * Atv(Atv(sAAo * x)) -> W3
    hipMemsetAsync(tmp, 0, featBytes, stream);
    hipMemsetAsync(hbuf, 0, featBytes, stream);
    SPMM(row, col, g2, x, tmp);                 // Atv(sAAo * x)
    SPMM(row, col, (const float*)nullptr, tmp, hbuf);  // Atv(.)
    GEMM(hbuf, g3, W3, c_3);

#undef SPMM
#undef GEMM
}
